// QuantizeEMA_39041252720882
// MI455X (gfx1250) — matrix operand order repaired and checked by reading, not compiled, not run on hardware
//
#include <hip/hip_runtime.h>
#include <hip/hip_bf16.h>

// ---------------------------------------------------------------------------
// VQ-VAE quantize + EMA update for MI455X (gfx1250, wave32, WMMA).
//
// Tokens T = 32*64*64 = 131072, C = 64, K = 512.
// Distance GEMM via v_wmma_f32_16x16x32_f16 (f16 in, f32 acc); argmin needs
// only |c|^2 - 2 x.c.  Codebook staged to LDS with async global->LDS loads.
// Each wave handles TWO 16-token tiles so every B fragment feeds 2 WMMAs.
// ---------------------------------------------------------------------------

typedef __attribute__((ext_vector_type(16))) _Float16 v16h;
typedef __attribute__((ext_vector_type(8)))  float    v8f;

#define T_TOKENS   131072
#define C_DIM      64
#define K_CODES    512
#define OUT_ELEMS  8388608      // 32*64*64*64
#define LDS_PITCH  72           // halves per codebook row in LDS (bank-spread pad)

union B16frag { uint4 u[2]; v16h h; };

extern __shared__ char smem[];

// --------------------------- kernel 0: prep --------------------------------
__global__ __launch_bounds__(256)
void k_prep(const float* __restrict__ cb, _Float16* __restrict__ cbh,
            float* __restrict__ cnorm)
{
    int k = blockIdx.x * 256 + threadIdx.x;
    if (k >= K_CODES) return;
    float s = 0.0f;
    #pragma unroll 8
    for (int c = 0; c < C_DIM; ++c) {
        float v = cb[k * C_DIM + c];
        s = fmaf(v, v, s);
        cbh[k * C_DIM + c] = (_Float16)v;
    }
    cnorm[k] = s;
}

// ---------------------- kernel 1: WMMA distances + argmin ------------------
// Block: 8 waves, 256 tokens (2 tiles of 16 per wave).  Codebook (f16) is
// staged into LDS via async global->LDS loads, overlapped with A loads.
__global__ __launch_bounds__(256)
void k_argmin(const float* __restrict__ x, const _Float16* __restrict__ cbh,
              const float* __restrict__ cnorm, int* __restrict__ idxbuf,
              float* __restrict__ counts)
{
    _Float16* cb_lds = (_Float16*)smem;                       // 512 rows * 72 halves
    float*    cn_lds = (float*)(smem + K_CODES * LDS_PITCH * 2);

    const int tid = threadIdx.x;

    // LDS byte offsets (low 32 bits of a generic LDS pointer == LDS offset).
    const unsigned cb_base = (unsigned)(unsigned long long)(void*)cb_lds;
    const unsigned cn_base = (unsigned)(unsigned long long)(void*)cn_lds;

    // --- async stage: codebook 64KB (packed 128B/row -> LDS pitch 144B) ---
    for (int i = tid; i < K_CODES * 8; i += 256) {
        int row = i >> 3, seg = i & 7;
        unsigned lds_addr = cb_base + (unsigned)(row * (LDS_PITCH * 2) + seg * 16);
        unsigned g_off    = (unsigned)(i * 16);
        asm volatile("global_load_async_to_lds_b128 %0, %1, %2"
                     :: "v"(lds_addr), "v"(g_off), "s"(cbh) : "memory");
    }
    // --- async stage: |c|^2 (2KB) ---
    for (int i = tid; i < K_CODES; i += 256) {
        unsigned lds_addr = cn_base + (unsigned)(i * 4);
        unsigned g_off    = (unsigned)(i * 4);
        asm volatile("global_load_async_to_lds_b32 %0, %1, %2"
                     :: "v"(lds_addr), "v"(g_off), "s"(cnorm) : "memory");
    }

    const int wave = tid >> 5;
    const int lane = tid & 31;
    const int col  = lane & 15;        // N (codeword-in-tile) / A row (token-in-tile)
    const int hs   = lane >> 4;        // half-wave select
    const int t0   = blockIdx.x * 256 + wave * 32;

    // ---- A fragments for two token tiles (overlaps with async staging) ----
    // 16-bit A layout: lane(M)=col; VGPR v holds a K pair; K split 0-7/16-23
    // (hs=0) vs 8-15/24-31 (hs=1).  xf[t][c] = x[(t>>12)*262144 + c*4096 + (t&4095)].
    const int tok0 = t0 + col;
    const int tok1 = t0 + 16 + col;
    const int xb0  = (tok0 >> 12) * 262144 + (tok0 & 4095);
    const int xb1  = (tok1 >> 12) * 262144 + (tok1 & 4095);

    v16h a0, a1, a2, a3;
    #pragma unroll
    for (int v = 0; v < 8; ++v) {
        int kp = (v >> 2) * 16 + hs * 8 + (v & 3) * 2;   // K index of pair base
        a0[2 * v]     = (_Float16)x[xb0 + (kp     ) * 4096];
        a0[2 * v + 1] = (_Float16)x[xb0 + (kp +  1) * 4096];
        a1[2 * v]     = (_Float16)x[xb0 + (kp + 32) * 4096];
        a1[2 * v + 1] = (_Float16)x[xb0 + (kp + 33) * 4096];
        a2[2 * v]     = (_Float16)x[xb1 + (kp     ) * 4096];
        a2[2 * v + 1] = (_Float16)x[xb1 + (kp +  1) * 4096];
        a3[2 * v]     = (_Float16)x[xb1 + (kp + 32) * 4096];
        a3[2 * v + 1] = (_Float16)x[xb1 + (kp + 33) * 4096];
    }

    // wait for async staging to land in LDS, then block barrier
    asm volatile("s_wait_asynccnt 0x0" ::: "memory");
    __syncthreads();

    float best0[8], best1[8];
    int   besti0[8], besti1[8];
    #pragma unroll
    for (int r = 0; r < 8; ++r) {
        best0[r] = 3.4e38f; besti0[r] = 0;
        best1[r] = 3.4e38f; besti1[r] = 0;
    }

    // ---- loop over 32 codeword tiles of 16; B reused by both token tiles ----
    for (int nt = 0; nt < 32; ++nt) {
        const int n = nt * 16 + col;
        // B (KxN) layout: K split 0-7/16-23 (hs=0) vs 8-15/24-31 (hs=1), N = col.
        const _Float16* brow = cb_lds + n * LDS_PITCH + hs * 8;
        B16frag b0, b1;
        b0.u[0] = ((const uint4*)brow)[0];
        b0.u[1] = ((const uint4*)brow)[2];
        b1.u[0] = ((const uint4*)(brow + 32))[0];
        b1.u[1] = ((const uint4*)(brow + 32))[2];

        v8f acc0 = {0.f, 0.f, 0.f, 0.f, 0.f, 0.f, 0.f, 0.f};
        v8f acc1 = {0.f, 0.f, 0.f, 0.f, 0.f, 0.f, 0.f, 0.f};
        // interleave the two accumulator chains to cover WMMA->WMMA RAW latency
        acc0 = __builtin_amdgcn_wmma_f32_16x16x32_f16(false, a0, false, b0.h,
                                                      (short)0, acc0, false, false);
        acc1 = __builtin_amdgcn_wmma_f32_16x16x32_f16(false, a2, false, b0.h,
                                                      (short)0, acc1, false, false);
        acc0 = __builtin_amdgcn_wmma_f32_16x16x32_f16(false, a1, false, b1.h,
                                                      (short)0, acc0, false, false);
        acc1 = __builtin_amdgcn_wmma_f32_16x16x32_f16(false, a3, false, b1.h,
                                                      (short)0, acc1, false, false);

        const float c2 = cn_lds[n];
        #pragma unroll
        for (int r = 0; r < 8; ++r) {
            float d0 = fmaf(-2.0f, acc0[r], c2);   // |c|^2 - 2 x.c
            if (d0 < best0[r]) { best0[r] = d0; besti0[r] = n; }
            float d1 = fmaf(-2.0f, acc1[r], c2);
            if (d1 < best1[r]) { best1[r] = d1; besti1[r] = n; }
        }
    }

    // ---- argmin across the 16 lanes of each half-wave (D: N = col) ----
    #pragma unroll
    for (int tt = 0; tt < 2; ++tt) {
        #pragma unroll
        for (int r = 0; r < 8; ++r) {
            float v  = tt ? best1[r]  : best0[r];
            int   bi = tt ? besti1[r] : besti0[r];
            #pragma unroll
            for (int off = 1; off < 16; off <<= 1) {
                float ov = __shfl_xor(v, off, 32);
                int   oi = __shfl_xor(bi, off, 32);
                if (ov < v || (ov == v && oi < bi)) { v = ov; bi = oi; }
            }
            // D row M = r + 8*hs  ->  token t0 + tt*16 + r + 8*hs
            if (col == r) {
                idxbuf[t0 + tt * 16 + r + hs * 8] = bi;
                atomicAdd(&counts[bi], 1.0f);
            }
        }
    }
}

// ------------------- kernel 2: gather q, write out, scatter sums -----------
__global__ __launch_bounds__(256)
void k_scatter(const float* __restrict__ x, const float* __restrict__ cb,
               const int* __restrict__ idxbuf, float* __restrict__ out,
               float* __restrict__ sums)
{
    int o = blockIdx.x * 256 + threadIdx.x;
    if (o >= OUT_ELEMS) return;
    // o = ((n*64 + c)*64 + h)*64 + w ; token t = n*4096 + h*64 + w
    int t = (o >> 18) * 4096 + (o & 4095);
    int c = (o >> 12) & 63;
    int k = idxbuf[t];
    float xv = x[o];
    float q  = cb[k * C_DIM + c];
    out[o] = xv + (q - xv);                 // straight-through estimator value
    atomicAdd(&sums[k * C_DIM + c], xv);
}

// -------------------------- kernel 3: EMA finalize -------------------------
__global__ __launch_bounds__(256)
void k_ema(const float* __restrict__ Ns, const float* __restrict__ ms,
           const float* __restrict__ counts, const float* __restrict__ sums,
           float* __restrict__ out_cb, float* __restrict__ out_N,
           float* __restrict__ out_m)
{
    const float G = 0.99f, OMG = 1.0f - 0.99f;
    int i = blockIdx.x * 256 + threadIdx.x;        // K*C = 32768
    if (i >= K_CODES * C_DIM) return;
    int k = i >> 6, c = i & 63;
    float cnt  = counts[k];
    bool  occ  = cnt > 0.0f;
    float Nold = Ns[k];
    float Nn   = occ ? Nold * G + cnt * OMG : Nold;
    float mo   = ms[i];
    float mn   = occ ? mo * G + sums[i] * OMG : mo;
    out_m[i]  = mn;
    out_cb[i] = mn / Nn;
    if (c == 0) out_N[k] = Nn;
}

// ------------------------------- launcher ----------------------------------
extern "C" void kernel_launch(void* const* d_in, const int* in_sizes, int n_in,
                              void* d_out, int out_size, void* d_ws, size_t ws_size,
                              hipStream_t stream)
{
    const float* x  = (const float*)d_in[0];   // [32,64,64,64]
    const float* cb = (const float*)d_in[1];   // [512,64]
    const float* Ns = (const float*)d_in[2];   // [512]
    const float* ms = (const float*)d_in[3];   // [512,64]
    float* out = (float*)d_out;

    // workspace layout
    char* ws = (char*)d_ws;
    _Float16* cbh    = (_Float16*)(ws);             // 65536 B
    float*    cnorm  = (float*)(ws + 65536);        //  2048 B
    float*    counts = (float*)(ws + 67584);        //  2048 B
    float*    sums   = (float*)(ws + 69632);        // 131072 B
    int*      idxb   = (int*)(ws + 200704);         // 524288 B

    // zero counts + sums (contiguous) every call — graph-capture-safe
    hipMemsetAsync(counts, 0, 2048 + 131072, stream);

    k_prep<<<2, 256, 0, stream>>>(cb, cbh, cnorm);

    size_t lds_bytes = (size_t)K_CODES * LDS_PITCH * 2 + K_CODES * 4;  // 75776
    k_argmin<<<T_TOKENS / 256, 256, lds_bytes, stream>>>(x, cbh, cnorm, idxb, counts);

    k_scatter<<<OUT_ELEMS / 256, 256, 0, stream>>>(x, cb, idxb, out, sums);

    float* out_cb = out + OUT_ELEMS;                 // 32768
    float* out_N  = out + OUT_ELEMS + 32768;         // 512
    float* out_m  = out + OUT_ELEMS + 32768 + 512;   // 32768
    k_ema<<<(K_CODES * C_DIM) / 256, 256, 0, stream>>>(Ns, ms, counts, sums,
                                                       out_cb, out_N, out_m);
}
